// RotaryAttention_76416058130572
// MI455X (gfx1250) — compile-verified, matrix operand order checked by reading, not run on hardware
//
#include <hip/hip_runtime.h>
#include <hip/hip_bf16.h>
#include <math.h>

// ---------------------------------------------------------------------------
// RotaryAttention for MI455X (gfx1250, wave32, WMMA).
// Phase 1: RoPE+f16 pack of Q/K, V transpose (all into d_ws).
// Phase 2: flash attention, S^T formulation, v_wmma_f32_16x16x32_f16.
// ---------------------------------------------------------------------------

typedef _Float16 half8 __attribute__((ext_vector_type(8)));
typedef _Float16 v16h  __attribute__((ext_vector_type(16)));
typedef float    v8f   __attribute__((ext_vector_type(8)));

#define NH      16
#define HD      64
#define SEQ     2048
#define NKV     2049          // 2048 keys + 1 register token
#define KPAD    2080          // padded to multiple of 32
#define BATCH   4

// 1/sqrt(64) * log2(e): scores come out in log2 domain -> native v_exp_f32
#define QSCALE  0.18033688011112042f
// log2(10000)/16
#define FREQC   0.83048202372184065f

__device__ __forceinline__ v8f wmma_f16(v16h a, v16h b, v8f c) {
    return __builtin_amdgcn_wmma_f32_16x16x32_f16(false, a, false, b,
                                                  (short)0, c, false, false);
}

__device__ __forceinline__ v16h cat8(half8 lo, half8 hi) {
    return __builtin_shufflevector(lo, hi, 0,1,2,3,4,5,6,7,8,9,10,11,12,13,14,15);
}

// Raw v_exp_f32: args are always <= 0 here; flush-to-zero below -126 is the
// desired flash-softmax semantics (avoids OCML's range-fix cmp/cndmask pairs).
__device__ __forceinline__ float fast_exp2(float x) {
    return __builtin_amdgcn_exp2f(x);
}

// ---------------------------------------------------------------------------
// Phase 1a: Q -> RoPE -> *QSCALE -> f16, layout [b][h][s][d]
// ---------------------------------------------------------------------------
__global__ __launch_bounds__(256) void rope_q_kernel(
    const float* __restrict__ qin, _Float16* __restrict__ Qh)
{
    size_t tid = (size_t)blockIdx.x * 256 + threadIdx.x;   // ((b*S+s)*NH+h)*HD+d
    int d = (int)(tid & 63);
    int h = (int)((tid >> 6) & 15);
    int s = (int)((tid >> 10) & 2047);
    int b = (int)(tid >> 21);

    float x = qin[tid];
    float val = x;
    if (d < 32) {
        int f = d & 15;
        float angle = (float)s * exp2f(-(float)f * FREQC);
        float sn, cs;
        __sincosf(angle, &sn, &cs);
        float other = (d < 16) ? -qin[tid + 16] : qin[tid - 16];
        val = x * cs + other * sn;
    }
    Qh[(((size_t)b * NH + h) * SEQ + s) * HD + d] = (_Float16)(val * QSCALE);
}

// ---------------------------------------------------------------------------
// Phase 1b: K -> RoPE -> f16, layout [b][h][key][d], key in [0,KPAD)
//   key==2048: register token (rotary half zeroed), key>2048: zero pad
// ---------------------------------------------------------------------------
__global__ __launch_bounds__(256) void rope_k_kernel(
    const float* __restrict__ kin, const float* __restrict__ kreg,
    _Float16* __restrict__ Kh)
{
    size_t tid = (size_t)blockIdx.x * 256 + threadIdx.x;   // [b][h][key][d]
    int d = (int)(tid & 63);
    size_t r1 = tid >> 6;
    int key = (int)(r1 % KPAD);
    size_t r2 = r1 / KPAD;
    int h = (int)(r2 & 15);
    int b = (int)(r2 >> 4);

    float val = 0.0f;
    if (key < SEQ) {
        size_t src = (((size_t)b * SEQ + key) * NH + h) * HD + d;
        float x = kin[src];
        val = x;
        if (d < 32) {
            int f = d & 15;
            float angle = (float)key * exp2f(-(float)f * FREQC);
            float sn, cs;
            __sincosf(angle, &sn, &cs);
            float other = (d < 16) ? -kin[src + 16] : kin[src - 16];
            val = x * cs + other * sn;
        }
    } else if (key == SEQ) {
        val = (d < 32) ? 0.0f : kreg[h * HD + d];   // reg_mask zeros rotary half
    }
    Kh[tid] = (_Float16)val;
}

// ---------------------------------------------------------------------------
// Phase 1c: V transpose -> f16, layout [b][h][d][key], key in [0,KPAD)
// LDS 16x16 tile transpose; grid = B*NH * 4(dtiles) * (KPAD/16)(keytiles)
// ---------------------------------------------------------------------------
__global__ __launch_bounds__(256) void vt_kernel(
    const float* __restrict__ vin, const float* __restrict__ vreg,
    _Float16* __restrict__ Vt)
{
    __shared__ float tile[16][17];
    int bidx = blockIdx.x;
    int kt = bidx % (KPAD / 16); bidx /= (KPAD / 16);
    int dt = bidx & 3;           bidx >>= 2;
    int h  = bidx & 15;
    int b  = bidx >> 4;

    int tx = threadIdx.x & 15, ty = threadIdx.x >> 4;
    int key = kt * 16 + ty;
    int d   = dt * 16 + tx;

    float v = 0.0f;
    if (key < SEQ)       v = vin[(((size_t)b * SEQ + key) * NH + h) * HD + d];
    else if (key == SEQ) v = vreg[h * HD + d];
    tile[ty][tx] = v;
    __syncthreads();

    // write d' = dt*16+ty, key' = kt*16+tx  (contiguous over tx)
    Vt[(((size_t)b * NH + h) * HD + dt * 16 + ty) * KPAD + kt * 16 + tx] =
        (_Float16)tile[tx][ty];
}

// ---------------------------------------------------------------------------
// Phase 2: flash attention. One wave = 16 query rows; block = 8 waves = 128 q.
// grid.x = B*NH*(SEQ/128).  S^T = K_tile x Q^T so softmax is per-lane and
// exp(S^T) feeds the PV WMMA A-operand with no cross-lane transpose.
// ---------------------------------------------------------------------------
__global__ __launch_bounds__(256) void fa_kernel(
    const _Float16* __restrict__ Qh, const _Float16* __restrict__ Kh,
    const _Float16* __restrict__ Vt, float* __restrict__ out)
{
    const int lane   = threadIdx.x & 31;
    const int wave   = threadIdx.x >> 5;
    const int wg     = blockIdx.x;
    const int qblk   = wg & 15;           // 16 blocks of 128 queries
    const int bh     = wg >> 4;           // b*NH + h
    const int b      = bh >> 4;
    const int h      = bh & 15;
    const int qbase  = qblk * 128 + wave * 16;
    const int lo     = lane & 15;
    const int hi8    = (lane & 16) ? 8 : 0;   // lane-half K/row offset
    const int dA     = hi8;                   // A/B slot pattern offset

    // ---- Q^T fragments: lane holds q-row qbase+lo, slots {dA..dA+7, dA+16..}
    const _Float16* qrow = Qh + ((size_t)bh * SEQ + (qbase + lo)) * HD;
    v16h qt0 = cat8(*(const half8*)(qrow +  0 + dA), *(const half8*)(qrow + 16 + dA));
    v16h qt1 = cat8(*(const half8*)(qrow + 32 + dA), *(const half8*)(qrow + 48 + dA));

    v8f o0 = {}, o1 = {}, o2 = {}, o3 = {};
    float m = -1e38f, lsum = 0.0f;

    const _Float16* kbase = Kh + (size_t)bh * KPAD * HD;
    const _Float16* vbase = Vt + (size_t)bh * HD * KPAD;

    for (int t = 0; t < KPAD / 32; ++t) {
        const int kb = t * 32;

        // ---- K A-fragments: lane = key row (kb+lo / kb+16+lo)
        const _Float16* kr0 = kbase + (size_t)(kb + lo) * HD;
        const _Float16* kr1 = kr0 + 16 * HD;
        v16h ka0c0 = cat8(*(const half8*)(kr0 +  0 + dA), *(const half8*)(kr0 + 16 + dA));
        v16h ka0c1 = cat8(*(const half8*)(kr0 + 32 + dA), *(const half8*)(kr0 + 48 + dA));
        v16h ka1c0 = cat8(*(const half8*)(kr1 +  0 + dA), *(const half8*)(kr1 + 16 + dA));
        v16h ka1c1 = cat8(*(const half8*)(kr1 + 32 + dA), *(const half8*)(kr1 + 48 + dA));

        // ---- S^T tiles (M=key, N=q), log2 domain (scale folded into Q)
        v8f c0 = {}; c0 = wmma_f16(ka0c0, qt0, c0); c0 = wmma_f16(ka0c1, qt1, c0);
        v8f c1 = {}; c1 = wmma_f16(ka1c0, qt0, c1); c1 = wmma_f16(ka1c1, qt1, c1);

        // ---- tail mask: keys >= NKV are padding
        if (kb + 32 > NKV) {
            #pragma unroll
            for (int r = 0; r < 8; ++r) {
                if (kb + r + hi8 >= NKV)      c0[r] = -1e30f;
                if (kb + 16 + r + hi8 >= NKV) c1[r] = -1e30f;
            }
        }

        // ---- online softmax (per-lane over 16 regs + one xor-16 shuffle)
        float mt = fmaxf(c0[0], c1[0]);
        #pragma unroll
        for (int r = 1; r < 8; ++r) { mt = fmaxf(mt, fmaxf(c0[r], c1[r])); }
        mt = fmaxf(mt, __shfl_xor(mt, 16));
        float mnew  = fmaxf(m, mt);
        float alpha = fast_exp2(m - mnew);

        union { v16h v; _Float16 e[16]; } pf;
        float ps = 0.0f;
        #pragma unroll
        for (int r = 0; r < 8; ++r) {
            float p0 = fast_exp2(c0[r] - mnew); ps += p0; pf.e[r]     = (_Float16)p0;
            float p1 = fast_exp2(c1[r] - mnew); ps += p1; pf.e[8 + r] = (_Float16)p1;
        }
        ps += __shfl_xor(ps, 16);
        lsum = lsum * alpha + ps;
        m = mnew;

        // ---- rescale O (row q = r + hi8 lives at lane r+hi8)
        #pragma unroll
        for (int r = 0; r < 8; ++r) {
            float sc = __shfl(alpha, r + hi8);
            o0[r] *= sc; o1[r] *= sc; o2[r] *= sc; o3[r] *= sc;
        }

        // ---- V B-fragments (rows of V^T) and PV accumulation
        const _Float16* vr0 = vbase + (size_t)(lo +  0) * KPAD + kb + dA;
        const _Float16* vr1 = vbase + (size_t)(lo + 16) * KPAD + kb + dA;
        const _Float16* vr2 = vbase + (size_t)(lo + 32) * KPAD + kb + dA;
        const _Float16* vr3 = vbase + (size_t)(lo + 48) * KPAD + kb + dA;
        v16h bv0 = cat8(*(const half8*)(vr0), *(const half8*)(vr0 + 16));
        v16h bv1 = cat8(*(const half8*)(vr1), *(const half8*)(vr1 + 16));
        v16h bv2 = cat8(*(const half8*)(vr2), *(const half8*)(vr2 + 16));
        v16h bv3 = cat8(*(const half8*)(vr3), *(const half8*)(vr3 + 16));
        o0 = wmma_f16(pf.v, bv0, o0);
        o1 = wmma_f16(pf.v, bv1, o1);
        o2 = wmma_f16(pf.v, bv2, o2);
        o3 = wmma_f16(pf.v, bv3, o3);
    }

    // ---- finalize: divide by softmax denominator, store (B,S,NH*HD)
    float invl = 1.0f / lsum;
    #pragma unroll
    for (int r = 0; r < 8; ++r) {
        float il = __shfl(invl, r + hi8);
        int q = qbase + r + hi8;
        size_t base = (((size_t)b * SEQ + q) * NH + h) * HD + lo;
        out[base +  0] = o0[r] * il;
        out[base + 16] = o1[r] * il;
        out[base + 32] = o2[r] * il;
        out[base + 48] = o3[r] * il;
    }
}

// ---------------------------------------------------------------------------
extern "C" void kernel_launch(void* const* d_in, const int* in_sizes, int n_in,
                              void* d_out, int out_size, void* d_ws, size_t ws_size,
                              hipStream_t stream)
{
    (void)in_sizes; (void)n_in; (void)out_size; (void)ws_size;

    const float* q    = (const float*)d_in[0];
    const float* k    = (const float*)d_in[1];
    const float* v    = (const float*)d_in[2];
    const float* kreg = (const float*)d_in[3];
    const float* vreg = (const float*)d_in[4];
    float* out = (float*)d_out;

    // workspace layout (f16)
    const size_t QH_ELEMS = (size_t)BATCH * NH * SEQ  * HD;   //  8,388,608
    const size_t KH_ELEMS = (size_t)BATCH * NH * KPAD * HD;   //  8,519,680
    _Float16* Qh = (_Float16*)d_ws;
    _Float16* Kh = Qh + QH_ELEMS;
    _Float16* Vt = Kh + KH_ELEMS;

    // Phase 1
    rope_q_kernel<<<(unsigned)(QH_ELEMS / 256), 256, 0, stream>>>(q, Qh);
    rope_k_kernel<<<(unsigned)(KH_ELEMS / 256), 256, 0, stream>>>(k, kreg, Kh);
    vt_kernel<<<BATCH * NH * 4 * (KPAD / 16), 256, 0, stream>>>(v, vreg, Vt);

    // Phase 2: B*NH*(SEQ/128) workgroups of 8 waves
    fa_kernel<<<BATCH * NH * (SEQ / 128), 256, 0, stream>>>(Qh, Kh, Vt, out);
}
